// Tokenizer_84396107366908
// MI455X (gfx1250) — matrix operand order repaired and checked by reading, not compiled, not run on hardware
//
#include <hip/hip_runtime.h>
#include <math.h>

#define B_   32
#define T_   4096
#define C_   128
#define V_   1024
#define ROWS (B_ * T_)
#define EPSF 1e-12f

typedef __attribute__((ext_vector_type(16))) __bf16          v16bf;
typedef __attribute__((ext_vector_type(8)))  float           v8f;
typedef __attribute__((ext_vector_type(4)))  float           v4f;
typedef __attribute__((ext_vector_type(8)))  unsigned short  ushort8;

union FragU { v16bf v; ushort8 h[2]; };

__device__ __forceinline__ unsigned short f32_to_bf16(float f) {
    unsigned u = __float_as_uint(f);
    u += 0x7FFFu + ((u >> 16) & 1u);   // round-to-nearest-even
    return (unsigned short)(u >> 16);
}

// ---------------- zero accumulators (graph-replay safe) ----------------
__global__ void k_zero(float* accs) {
    if (threadIdx.x < 3) accs[threadIdx.x] = 0.0f;   // [0]=smooth [1]=commit [2]=masksum
}

// ---------------- codebook prep: bf16 convert + ||e||^2 ----------------
__global__ void k_prep_e(const float* __restrict__ e_full,
                         float* __restrict__ e_norm2,
                         unsigned short* __restrict__ e_bf) {
    __shared__ float ss[128];
    int row = blockIdx.x;            // 0..1023
    int k   = threadIdx.x;           // 0..127
    float v = e_full[(row + 1) * C_ + k];   // skip index 0 of e_full
    ss[k] = v * v;
    __syncthreads();
    for (int s = 64; s > 0; s >>= 1) {
        if (k < s) ss[k] += ss[k + s];
        __syncthreads();
    }
    if (k == 0) e_norm2[row] = ss[0];
    e_bf[row * C_ + k] = f32_to_bf16(v);
}

// ---------------- sum(mask) ----------------
__global__ void k_mask_sum(const float* __restrict__ mask, float* accs) {
    __shared__ float ss[256];
    int i = blockIdx.x * 256 + threadIdx.x;
    ss[threadIdx.x] = mask[i];
    __syncthreads();
    for (int s = 128; s > 0; s >>= 1) {
        if (threadIdx.x < s) ss[threadIdx.x] += ss[threadIdx.x + s];
        __syncthreads();
    }
    if (threadIdx.x == 0) atomicAdd(&accs[2], ss[0]);
}

// ---------------- main fused kernel ----------------
// One WG = 16 rows x full V=1024 columns.  8 waves; wave w owns columns [w*128, w*128+128).
__global__ __launch_bounds__(256) void k_main(
    const float* __restrict__ z, const float* __restrict__ mask,
    const float* __restrict__ e_full, const float* __restrict__ e_norm2,
    const unsigned short* __restrict__ e_bf,
    float* __restrict__ lp_out, float* __restrict__ zq_out, float* accs)
{
    __shared__ float          zn[16 * C_];        // normalized z, f32
    __shared__ unsigned short abf[16 * 136];      // normalized z, bf16, padded stride
    __shared__ float          red[256];
    __shared__ float          rowinv[16], rownrm2[16];
    __shared__ float          red_min[8][16];
    __shared__ int            red_idx[8][16];
    __shared__ float          red_sum[8][16];
    __shared__ float          rowmin[16], rowlse[16];
    __shared__ int            rowidx[16];

    const int  tid  = threadIdx.x;
    const int  wave = tid >> 5;
    const int  lane = tid & 31;
    const long R0   = (long)blockIdx.x * 16;

    // ---- Phase 0: load 16x128 z tile (b128 loads), normalize ----
    const int lrow = tid >> 4;          // 0..15
    const int lk   = (tid & 15) * 8;    // 0..120
    float v[8];
    {
        const float* zp = z + (R0 + lrow) * C_ + lk;
        v4f a = *(const v4f*)(zp);
        v4f b = *(const v4f*)(zp + 4);
        #pragma unroll
        for (int i = 0; i < 4; i++) { v[i] = a[i]; v[4 + i] = b[i]; }
    }
    float ssl = 0.0f;
    #pragma unroll
    for (int i = 0; i < 8; i++) ssl += v[i] * v[i];
    red[tid] = ssl;
    __syncthreads();
    if (tid < 16) {
        float s = 0.0f;
        #pragma unroll
        for (int j = 0; j < 16; j++) s += red[tid * 16 + j];
        float inv = 1.0f / fmaxf(sqrtf(s), EPSF);
        rowinv[tid]  = inv;
        rownrm2[tid] = s * inv * inv;   // == sum(z_norm^2), ~1
    }
    __syncthreads();
    {
        float inv = rowinv[lrow];
        #pragma unroll
        for (int i = 0; i < 8; i++) {
            float zv = v[i] * inv;
            zn[lrow * C_ + lk + i]   = zv;
            abf[lrow * 136 + lk + i] = f32_to_bf16(zv);
        }
    }
    __syncthreads();

    // ---- Phase 1: WMMA GEMM d = ||z||^2 - 2 z.e + ||e||^2, min/argmin on the fly ----
    const int half = lane >> 4;
    const int l16  = lane & 15;

    FragU afrag[4];
    #pragma unroll
    for (int ks = 0; ks < 4; ks++) {
        // A 16x32 bf16: lanes 0-15 hold K0-7 & K16-23, lanes 16-31 hold K8-15 & K24-31
        const unsigned short* ap = &abf[l16 * 136 + ks * 32 + half * 8];
        afrag[ks].h[0] = *(const ushort8*)(ap);
        afrag[ks].h[1] = *(const ushort8*)(ap + 16);
    }

    const int NB = wave * 128;
    float dreg[8][8];
    float minv[8];
    int   mini[8];
    #pragma unroll
    for (int r = 0; r < 8; r++) { minv[r] = 3.402823466e38f; mini[r] = 0; }

    #pragma unroll
    for (int nt = 0; nt < 8; nt++) {
        const int ncol = NB + nt * 16 + l16;   // this lane's column for B / C / D
        v8f acc = {0.f, 0.f, 0.f, 0.f, 0.f, 0.f, 0.f, 0.f};
        #pragma unroll
        for (int ks = 0; ks < 4; ks++) {
            // B 32x16 bf16: lanes 0-15 hold K0-7 & K16-23, lanes 16-31 hold K8-15 & K24-31
            FragU bfrag;
            const unsigned short* bp = e_bf + ncol * C_ + ks * 32 + half * 8;
            bfrag.h[0] = *(const ushort8*)(bp);
            bfrag.h[1] = *(const ushort8*)(bp + 16);
            acc = __builtin_amdgcn_wmma_f32_16x16x32_bf16(
                false, afrag[ks].v, false, bfrag.v, (short)0, acc, false, false);
        }
        const float en2 = e_norm2[ncol];
        #pragma unroll
        for (int r = 0; r < 8; r++) {
            const int m = r + half * 8;
            float d = rownrm2[m] - 2.0f * acc[r] + en2;
            dreg[nt][r] = d;
            if (d < minv[r] || (d == minv[r] && ncol < mini[r])) { minv[r] = d; mini[r] = ncol; }
        }
    }

    // min/argmin across the 16-lane column group
    #pragma unroll
    for (int r = 0; r < 8; r++) {
        float mv = minv[r]; int mi = mini[r];
        #pragma unroll
        for (int s = 1; s < 16; s <<= 1) {
            float ov = __shfl_xor(mv, s, 32);
            int   oi = __shfl_xor(mi, s, 32);
            if (ov < mv || (ov == mv && oi < mi)) { mv = ov; mi = oi; }
        }
        minv[r] = mv; mini[r] = mi;
    }
    if (l16 == 0) {
        #pragma unroll
        for (int r = 0; r < 8; r++) {
            red_min[wave][r + half * 8] = minv[r];
            red_idx[wave][r + half * 8] = mini[r];
        }
    }
    __syncthreads();
    if (tid < 16) {
        float mv = red_min[0][tid]; int mi = red_idx[0][tid];
        #pragma unroll
        for (int w = 1; w < 8; w++) {
            float ov = red_min[w][tid]; int oi = red_idx[w][tid];
            if (ov < mv || (ov == mv && oi < mi)) { mv = ov; mi = oi; }
        }
        rowmin[tid] = mv; rowidx[tid] = mi;
    }
    __syncthreads();

    // ---- Phase 2: logsumexp(-d) per row ----
    #pragma unroll
    for (int r = 0; r < 8; r++) {
        const int m = r + half * 8;
        const float rm = rowmin[m];
        float s = 0.0f;
        #pragma unroll
        for (int nt = 0; nt < 8; nt++) s += __expf(rm - dreg[nt][r]);
        #pragma unroll
        for (int sh = 1; sh < 16; sh <<= 1) s += __shfl_xor(s, sh, 32);
        if (l16 == 0) red_sum[wave][m] = s;
    }
    __syncthreads();
    if (tid < 16) {
        float s = 0.0f;
        #pragma unroll
        for (int w = 0; w < 8; w++) s += red_sum[w][tid];
        rowlse[tid] = __logf(s);
    }
    __syncthreads();

    // ---- Phase 3: stream out log_probs (write-once: non-temporal, keep L2 for e/z) ----
    #pragma unroll
    for (int nt = 0; nt < 8; nt++) {
        const int ncol = NB + nt * 16 + l16;
        #pragma unroll
        for (int r = 0; r < 8; r++) {
            const int m = r + half * 8;
            float lp = (rowmin[m] - dreg[nt][r]) - rowlse[m];
            __builtin_nontemporal_store(lp, &lp_out[(R0 + m) * (long)V_ + ncol]);
        }
    }

    // ---- Phase 4: z_q gather (f32 codebook) + commitment loss partial ----
    float csum = 0.0f;
    {
        const int   row = tid >> 4;
        const int   k0  = (tid & 15) * 8;
        const int   idx = rowidx[row];
        const float mk  = mask[R0 + row];
        const float* ep = e_full + (long)(idx + 1) * C_ + k0;
        float* zq = zq_out + (R0 + row) * (long)C_ + k0;
        v4f q0, q1;
        #pragma unroll
        for (int i = 0; i < 4; i++) {
            q0[i] = ep[i] * mk;
            q1[i] = ep[4 + i] * mk;
        }
        __builtin_nontemporal_store(q0, (v4f*)(zq));
        __builtin_nontemporal_store(q1, (v4f*)(zq + 4));
        #pragma unroll
        for (int i = 0; i < 4; i++) {
            float d0 = zn[row * C_ + k0 + i]     - q0[i];
            float d1 = zn[row * C_ + k0 + 4 + i] - q1[i];
            csum += d0 * d0 * mk + d1 * d1 * mk;
        }
    }
    red[tid] = csum;
    __syncthreads();
    if (tid < 32) {
        float s = 0.0f;
        #pragma unroll
        for (int j = 0; j < 8; j++) s += red[tid + 32 * j];
        red[tid] = s;
    }
    __syncthreads();
    if (tid == 0) {
        float s = 0.0f;
        #pragma unroll
        for (int j = 0; j < 32; j++) s += red[j];
        atomicAdd(&accs[1], s);
    }
}

// ---------------- smoothness loss: sum((zn[t]-zn[t+1])^2 * mask[t+1]) ----------------
__global__ void k_smooth(const float* __restrict__ z, const float* __restrict__ mask,
                         float* accs) {
    __shared__ float wsum[8];
    const int wave = threadIdx.x >> 5;
    const int lane = threadIdx.x & 31;
    const long row = (long)blockIdx.x * 8 + wave;
    float partial = 0.0f;
    if ((row % T_) != (T_ - 1)) {        // wave-uniform branch
        const float* a = z + row * C_ + lane * 4;
        v4f av = *(const v4f*)(a);
        v4f bv = *(const v4f*)(a + C_);
        float s0 = 0.0f, s1 = 0.0f;
        #pragma unroll
        for (int i = 0; i < 4; i++) { s0 += av[i] * av[i]; s1 += bv[i] * bv[i]; }
        #pragma unroll
        for (int sh = 1; sh < 32; sh <<= 1) {
            s0 += __shfl_xor(s0, sh, 32);
            s1 += __shfl_xor(s1, sh, 32);
        }
        const float i0 = 1.0f / fmaxf(sqrtf(s0), EPSF);
        const float i1 = 1.0f / fmaxf(sqrtf(s1), EPSF);
        float ds = 0.0f;
        #pragma unroll
        for (int i = 0; i < 4; i++) {
            float d = av[i] * i0 - bv[i] * i1;
            ds += d * d;
        }
        #pragma unroll
        for (int sh = 1; sh < 32; sh <<= 1) ds += __shfl_xor(ds, sh, 32);
        partial = ds * mask[row + 1];
    }
    if (lane == 0) wsum[wave] = partial;
    __syncthreads();
    if (threadIdx.x == 0) {
        float s = 0.0f;
        #pragma unroll
        for (int w = 0; w < 8; w++) s += wsum[w];
        atomicAdd(&accs[0], s);
    }
}

// ---------------- finalize scalars ----------------
__global__ void k_final(const float* accs, float* out) {
    if (threadIdx.x == 0) {
        float vc = accs[2] * (float)C_;   // sum(mask) * C
        out[0] = accs[0] / vc;            // smoothness_loss
        out[1] = accs[1] / vc;            // commitment_loss
    }
}

extern "C" void kernel_launch(void* const* d_in, const int* in_sizes, int n_in,
                              void* d_out, int out_size, void* d_ws, size_t ws_size,
                              hipStream_t stream) {
    const float* z      = (const float*)d_in[0];   // [32,4096,128]
    const float* mask   = (const float*)d_in[1];   // [32,4096,1]
    const float* e_full = (const float*)d_in[2];   // [1025,128]
    float* out = (float*)d_out;

    char* ws = (char*)d_ws;
    float*          accs    = (float*)ws;                         // 3 accumulators
    float*          e_norm2 = (float*)(ws + 256);                 // 1024 f32
    unsigned short* e_bf    = (unsigned short*)(ws + 256 + 4096); // 1024x128 bf16

    float* lp_out = out + 2;                                      // [32,4096,1024]
    float* zq_out = out + 2 + (long)ROWS * V_;                    // [32,4096,128]

    k_zero    <<<1,         32,  0, stream>>>(accs);
    k_prep_e  <<<V_,        128, 0, stream>>>(e_full, e_norm2, e_bf);
    k_mask_sum<<<ROWS/256,  256, 0, stream>>>(mask, accs);
    k_main    <<<ROWS/16,   256, 0, stream>>>(z, mask, e_full, e_norm2, e_bf,
                                              lp_out, zq_out, accs);
    k_smooth  <<<ROWS/8,    256, 0, stream>>>(z, mask, accs);
    k_final   <<<1,         32,  0, stream>>>(accs, out);
}
